// Model_54176717471980
// MI455X (gfx1250) — matrix operand order repaired and checked by reading, not compiled, not run on hardware
//
#include <hip/hip_runtime.h>
#include <math.h>

// ---- Problem dimensions (fixed by the reference) ----
#define BDIM 64
#define TDIM 512
#define IDIM 256
#define HDIM 1024
#define GDIM 4096   // 4*H

typedef __attribute__((ext_vector_type(16))) __bf16 bf16x16;
typedef __attribute__((ext_vector_type(8)))  float  f32x8;

union TileAB {
    bf16x16 v;
    uint4   q[2];
};

// float -> bf16 with round-to-nearest-even (bit-level, avoids relying on __bf16 scalar conversions)
__device__ __forceinline__ unsigned short f2bf(float f) {
    unsigned int u = __float_as_uint(f);
    unsigned int r = u + 0x7FFFu + ((u >> 16) & 1u);
    return (unsigned short)(r >> 16);
}

__device__ __forceinline__ float sigmoidf_(float x) {
    return 1.0f / (1.0f + __expf(-x));
}

// ---------------- Prep kernels ----------------
__global__ void cvt_f32_bf16_kernel(const float* __restrict__ src,
                                    unsigned short* __restrict__ dst, int n) {
    int i = blockIdx.x * blockDim.x + threadIdx.x;
    if (i < n) dst[i] = f2bf(src[i]);
}

__global__ void bias_sum_kernel(const float* __restrict__ b_ih,
                                const float* __restrict__ b_hh,
                                float* __restrict__ bias) {
    int i = blockIdx.x * blockDim.x + threadIdx.x;
    if (i < GDIM) bias[i] = b_ih[i] + b_hh[i];
}

__global__ void init_state_kernel(float* __restrict__ hF, float* __restrict__ cF,
                                  unsigned short* __restrict__ hbf) {
    int i = blockIdx.x * blockDim.x + threadIdx.x;
    if (i < BDIM * HDIM) { hF[i] = 0.0f; cF[i] = 0.0f; hbf[i] = 0; }
}

// ---------------- Per-step gates GEMM (WMMA bf16 -> f32) ----------------
// gates[b, n] = sum_k xbf[b, t, k] * Wih[n, k]  (k < IDIM)
//            +  sum_k hbf[b, k]    * Whh[n, k]  (k < HDIM)
//            +  bias[n]
// One wave computes one 16x16 tile of the [64, 4096] gate matrix.
// 1024 tiles total -> 256 blocks x 4 waves (128 threads).
__global__ __launch_bounds__(128)
void lstm_gates_kernel(const unsigned short* __restrict__ xbf,   // [B, T, I] bf16
                       int t,
                       const unsigned short* __restrict__ hbf,   // [B, H] bf16
                       const unsigned short* __restrict__ WihBf, // [4H, I] bf16
                       const unsigned short* __restrict__ WhhBf, // [4H, H] bf16
                       const float* __restrict__ bias,           // [4H]
                       float* __restrict__ gates)                // [B, 4H]
{
    const int wave = threadIdx.x >> 5;
    const int lane = threadIdx.x & 31;
    const int tile = blockIdx.x * 4 + wave;   // 0..1023
    const int mt   = tile & 3;                // batch tile  (4)
    const int nt   = tile >> 2;               // gate tile   (256)
    const int m0   = mt * 16;
    const int n0   = nt * 16;
    const int half = lane >> 4;               // 0 or 1
    const int l16  = lane & 15;
    const int mrow = m0 + l16;                // A-row this lane loads
    const int ncol = n0 + l16;                // B-col this lane loads

    f32x8 acc = {};

    // ---- x_t part: K = IDIM ----
    const unsigned short* arow = xbf + ((size_t)mrow * TDIM + (size_t)t) * IDIM;
    const unsigned short* brow = WihBf + (size_t)ncol * IDIM;
    for (int k = 0; k < IDIM; k += 32) {
        TileAB a, b;
        a.q[0] = *(const uint4*)(arow + k + half * 8);
        a.q[1] = *(const uint4*)(arow + k + 16 + half * 8);
        b.q[0] = *(const uint4*)(brow + k + half * 8);
        b.q[1] = *(const uint4*)(brow + k + 16 + half * 8);
        acc = __builtin_amdgcn_wmma_f32_16x16x32_bf16(
            false, a.v, false, b.v, (short)0, acc, false, false);
    }

    // ---- h part: K = HDIM ----
    arow = hbf + (size_t)mrow * HDIM;
    brow = WhhBf + (size_t)ncol * HDIM;
    __builtin_prefetch(brow, 0, 1);           // global_prefetch_b8 (weights live in L2)
    for (int k = 0; k < HDIM; k += 32) {
        TileAB a, b;
        a.q[0] = *(const uint4*)(arow + k + half * 8);
        a.q[1] = *(const uint4*)(arow + k + 16 + half * 8);
        b.q[0] = *(const uint4*)(brow + k + half * 8);
        b.q[1] = *(const uint4*)(brow + k + 16 + half * 8);
        acc = __builtin_amdgcn_wmma_f32_16x16x32_bf16(
            false, a.v, false, b.v, (short)0, acc, false, false);
    }

    // ---- write tile + bias. C/D layout: VGPR r -> row m0 + r + 8*half, col n0 + l16
    const float bn = bias[ncol];
    float* gout = gates + n0 + l16;
#pragma unroll
    for (int r = 0; r < 8; ++r) {
        int m = m0 + r + 8 * half;
        gout[(size_t)m * GDIM] = acc[r] + bn;
    }
}

// ---------------- Per-step cell update ----------------
__global__ __launch_bounds__(256)
void lstm_cell_kernel(const float* __restrict__ gates,   // [B, 4H]
                      float* __restrict__ hF,            // [B, H]
                      float* __restrict__ cF,            // [B, H]
                      unsigned short* __restrict__ hbf)  // [B, H] bf16
{
    int idx = blockIdx.x * blockDim.x + threadIdx.x;     // 0 .. B*H-1
    if (idx >= BDIM * HDIM) return;
    int b = idx >> 10;          // /HDIM
    int j = idx & (HDIM - 1);
    const float* g = gates + (size_t)b * GDIM;
    float ig = sigmoidf_(g[j]);
    float fg = sigmoidf_(g[HDIM + j]);
    float gg = tanhf(g[2 * HDIM + j]);
    float og = sigmoidf_(g[3 * HDIM + j]);
    float cn = fg * cF[idx] + ig * gg;
    float hn = og * tanhf(cn);
    cF[idx] = cn;
    hF[idx] = hn;
    hbf[idx] = f2bf(hn);
}

// ---------------- Final: out[b] = tanh(h) . fc_W + fc_b ----------------
__global__ __launch_bounds__(256)
void lstm_final_kernel(const float* __restrict__ hF,
                       const float* __restrict__ fcW,
                       const float* __restrict__ fcb,
                       float* __restrict__ out)
{
    __shared__ float red[256];
    int b = blockIdx.x;
    float s = 0.0f;
    for (int j = threadIdx.x; j < HDIM; j += 256)
        s += tanhf(hF[(size_t)b * HDIM + j]) * fcW[j];
    red[threadIdx.x] = s;
    __syncthreads();
    for (int off = 128; off > 0; off >>= 1) {
        if (threadIdx.x < off) red[threadIdx.x] += red[threadIdx.x + off];
        __syncthreads();
    }
    if (threadIdx.x == 0) out[b] = red[0] + fcb[0];
}

// ---------------- Host launch ----------------
extern "C" void kernel_launch(void* const* d_in, const int* in_sizes, int n_in,
                              void* d_out, int out_size, void* d_ws, size_t ws_size,
                              hipStream_t stream) {
    const float* x    = (const float*)d_in[0];   // [B, T, I]
    const float* W_ih = (const float*)d_in[1];   // [4H, I]
    const float* W_hh = (const float*)d_in[2];   // [4H, H]
    const float* b_ih = (const float*)d_in[3];   // [4H]
    const float* b_hh = (const float*)d_in[4];   // [4H]
    const float* fc_W = (const float*)d_in[5];   // [O, H], O=1
    const float* fc_b = (const float*)d_in[6];   // [O]
    float* out = (float*)d_out;                  // [B, O] = [64]

    // ---- workspace carve (all offsets 256B-aligned by construction) ----
    char* p = (char*)d_ws;
    unsigned short* xbf   = (unsigned short*)p; p += (size_t)BDIM * TDIM * IDIM * 2; // 16 MB
    unsigned short* WihBf = (unsigned short*)p; p += (size_t)GDIM * IDIM * 2;        //  2 MB
    unsigned short* WhhBf = (unsigned short*)p; p += (size_t)GDIM * HDIM * 2;        //  8 MB
    float*          bias  = (float*)p;          p += (size_t)GDIM * 4;
    float*          hF    = (float*)p;          p += (size_t)BDIM * HDIM * 4;
    float*          cF    = (float*)p;          p += (size_t)BDIM * HDIM * 4;
    unsigned short* hbf   = (unsigned short*)p; p += (size_t)BDIM * HDIM * 2;
    float*          gates = (float*)p;          p += (size_t)BDIM * GDIM * 4;
    (void)ws_size; (void)in_sizes; (void)n_in; (void)out_size;

    // ---- prep ----
    {
        int n = BDIM * TDIM * IDIM;
        cvt_f32_bf16_kernel<<<(n + 255) / 256, 256, 0, stream>>>(x, xbf, n);
    }
    {
        int n = GDIM * IDIM;
        cvt_f32_bf16_kernel<<<(n + 255) / 256, 256, 0, stream>>>(W_ih, WihBf, n);
    }
    {
        int n = GDIM * HDIM;
        cvt_f32_bf16_kernel<<<(n + 255) / 256, 256, 0, stream>>>(W_hh, WhhBf, n);
    }
    bias_sum_kernel<<<(GDIM + 255) / 256, 256, 0, stream>>>(b_ih, b_hh, bias);
    init_state_kernel<<<(BDIM * HDIM + 255) / 256, 256, 0, stream>>>(hF, cF, hbf);

    // ---- recurrence: 512 sequential steps, 2 kernels per step ----
    for (int t = 0; t < TDIM; ++t) {
        lstm_gates_kernel<<<256, 128, 0, stream>>>(xbf, t, hbf, WihBf, WhhBf, bias, gates);
        lstm_cell_kernel<<<(BDIM * HDIM + 255) / 256, 256, 0, stream>>>(gates, hF, cF, hbf);
    }

    // ---- final projection ----
    lstm_final_kernel<<<BDIM, 256, 0, stream>>>(hF, fc_W, fc_b, out);
}
